// MADE_1803886264390
// MI455X (gfx1250) — matrix operand order repaired and checked by reading, not compiled, not run on hardware
//
#include <hip/hip_runtime.h>

typedef __attribute__((ext_vector_type(16))) __bf16 v16bf;
typedef __attribute__((ext_vector_type(8)))  float  v8f;

// f32 -> bf16 (round to nearest even), as raw bits
__device__ __forceinline__ unsigned short f2bf(float f) {
    unsigned u = __builtin_bit_cast(unsigned, f);
    unsigned r = (u + 0x7FFFu + ((u >> 16) & 1u)) >> 16;
    return (unsigned short)r;
}
__device__ __forceinline__ float bf2f(unsigned short h) {
    return __builtin_bit_cast(float, ((unsigned)h) << 16);
}

// Pack masked weights into WMMA per-lane operand layouts (bf16).
//   w0p: [256][512] halfs, A-side pack order (kt*32 + g*16 + e), column i of mw0
//   w1p: [32*16 tiles][512] halfs, B operand tiles for layer-1
//   w2p: [512][512] halfs, plain row-major masked mw2
__global__ void made_prep(const float* __restrict__ w0,
                          const float* __restrict__ w1,
                          const float* __restrict__ w2,
                          unsigned short* __restrict__ ws) {
    int id = blockIdx.x * blockDim.x + threadIdx.x;
    unsigned short* w0p = ws;
    unsigned short* w1p = ws + 131072;   // 256*512
    unsigned short* w2p = ws + 393216;   // + 512*512
    if (id < 131072) {
        int i = id >> 9, p = id & 511;
        int kt = p >> 5, rem = p & 31, g = rem >> 4, e = rem & 15;
        int j = kt * 32 + (e + (g << 3) + (e & 8));          // A-layout K map
        float v = ((j % 255) >= i) ? w0[j * 256 + i] : 0.f;  // m0 = d_h[j] >= d_in[i]
        w0p[id] = f2bf(v);
    } else if (id < 393216) {
        int q = id - 131072;
        int tb = q >> 9, r = q & 511;
        int jt = tb >> 4, kt = tb & 15;
        int l = r >> 4, e = r & 15;
        int j = jt * 16 + (l & 15);
        int k = kt * 32 + (e + ((l >> 4) << 3) + (e & 8));   // B-layout K map
        float v = ((j % 255) >= (k % 255)) ? w1[j * 512 + k] : 0.f;  // m1
        w1p[q] = f2bf(v);
    } else if (id < 655360) {
        int q = id - 393216;
        int o = q >> 9, k = q & 511;
        float v = (((o & 255) - 1) >= (k % 255)) ? w2[o * 512 + k] : 0.f;  // m2 tiled
        w2p[q] = f2bf(v);
    }
}

// One workgroup = 16 batch rows, 8 waves, runs the full 256-step scan.
__global__ void __launch_bounds__(256, 1)
made_main(const float* __restrict__ z,  const float* __restrict__ b0,
          const float* __restrict__ b1, const float* __restrict__ b2,
          const unsigned short* __restrict__ w0p,
          const unsigned short* __restrict__ w1p,
          const unsigned short* __restrict__ w2p,
          float* __restrict__ out) {
    __shared__ __align__(32) unsigned short h0p[16 * 512];  // packed A tiles (bf16), 16KB
    __shared__ float h1p[16 * 513];                         // padded to kill bank conflicts
    __shared__ float xs[16];
    __shared__ __align__(16) unsigned short w0ls[512];      // async-staged w0 column
    __shared__ float red[8 * 32];

    const int t = threadIdx.x;
    const int w = t >> 5, l = t & 31;
    const int m = l & 15, g = l >> 4;
    const int wg = blockIdx.x;

    // a0 state in registers: thread owns k-tiles {2w, 2w+1}; init with b0 (x==0).
    float a0[2][16];
#pragma unroll
    for (int tt = 0; tt < 2; ++tt) {
        int kt = 2 * w + tt;
#pragma unroll
        for (int e = 0; e < 16; ++e)
            a0[tt][e] = b0[kt * 32 + (e + (g << 3) + (e & 8))];
    }
    float ld = 0.f;

    const unsigned long long w0base = (unsigned long long)w0p;
    const unsigned ldsbase = (unsigned)(size_t)(&w0ls[0]);

    for (int i = 0; i < 256; ++i) {
        // ---- async prefetch of w0 column i into LDS (consumed at step end) ----
        if (w == 0) {
            unsigned voff = (unsigned)(i * 1024 + l * 16);
            asm volatile("global_load_async_to_lds_b128 %0, %1, %2"
                         :: "v"(ldsbase + (unsigned)(l * 16)), "v"(voff), "s"(w0base)
                         : "memory");
            asm volatile("global_load_async_to_lds_b128 %0, %1, %2"
                         :: "v"(ldsbase + (unsigned)(l * 16 + 512)), "v"(voff + 512u), "s"(w0base)
                         : "memory");
        }
        // ---- h0 = relu(a0) -> packed bf16 A tiles in LDS ----
#pragma unroll
        for (int tt = 0; tt < 2; ++tt) {
            unsigned pk[8];
#pragma unroll
            for (int p = 0; p < 8; ++p) {
                float f0 = fmaxf(a0[tt][2 * p], 0.f);
                float f1 = fmaxf(a0[tt][2 * p + 1], 0.f);
                pk[p] = (unsigned)f2bf(f0) | ((unsigned)f2bf(f1) << 16);
            }
            uint4* dst = (uint4*)&h0p[(2 * w + tt) * 512 + l * 16];
            dst[0] = make_uint4(pk[0], pk[1], pk[2], pk[3]);
            dst[1] = make_uint4(pk[4], pk[5], pk[6], pk[7]);
        }
        __syncthreads();

        // ---- layer 1: wave w computes output j-tiles {4w..4w+3}, K = 512 ----
        v8f acc0 = {}, acc1 = {}, acc2 = {}, acc3 = {};
#pragma unroll 4
        for (int kt = 0; kt < 16; ++kt) {
            v16bf A = *(const v16bf*)&h0p[kt * 512 + l * 16];
            const int jb = 4 * w;
            v16bf B0 = *(const v16bf*)&w1p[((jb + 0) * 16 + kt) * 512 + l * 16];
            v16bf B1 = *(const v16bf*)&w1p[((jb + 1) * 16 + kt) * 512 + l * 16];
            v16bf B2 = *(const v16bf*)&w1p[((jb + 2) * 16 + kt) * 512 + l * 16];
            v16bf B3 = *(const v16bf*)&w1p[((jb + 3) * 16 + kt) * 512 + l * 16];
            acc0 = __builtin_amdgcn_wmma_f32_16x16x32_bf16(false, A, false, B0, (short)0, acc0, false, false);
            acc1 = __builtin_amdgcn_wmma_f32_16x16x32_bf16(false, A, false, B1, (short)0, acc1, false, false);
            acc2 = __builtin_amdgcn_wmma_f32_16x16x32_bf16(false, A, false, B2, (short)0, acc2, false, false);
            acc3 = __builtin_amdgcn_wmma_f32_16x16x32_bf16(false, A, false, B3, (short)0, acc3, false, false);
        }
        // bias + relu -> h1 plain layout in LDS (C/D layout: lane n=l&15, M = r + g*8)
        {
            int n = l & 15;
#pragma unroll
            for (int q = 0; q < 4; ++q) {
                int jt = 4 * w + q;
                float bias = b1[jt * 16 + n];
                v8f acc = (q == 0) ? acc0 : (q == 1) ? acc1 : (q == 2) ? acc2 : acc3;
#pragma unroll
                for (int r = 0; r < 8; ++r)
                    h1p[(r + g * 8) * 513 + jt * 16 + n] = fmaxf(acc[r] + bias, 0.f);
            }
        }
        __syncthreads();

        // ---- layer 2: 32 dots (16 batch x {mu,alpha}) split-K over 8 waves ----
        {
            int row = (g ? 256 : 0) + i;
            const unsigned short* w2r = &w2p[row * 512 + w * 64];
            const float* h1r = &h1p[m * 513 + w * 64];
            float part = 0.f;
#pragma unroll 8
            for (int jj = 0; jj < 64; ++jj)
                part += h1r[jj] * bf2f(w2r[jj]);
            red[w * 32 + l] = part;
        }
        __syncthreads();

        // ---- reduce + autoregressive update (wave 0) ----
        if (w == 0) {
            float s = 0.f;
#pragma unroll
            for (int p = 0; p < 8; ++p) s += red[p * 32 + l];
            s += b2[(g ? 256 : 0) + i];
            float alpha = __shfl(s, m + 16, 32);
            float mu    = __shfl(s, m, 32);
            if (g == 0) {
                float xb = mu + z[(wg * 16 + m) * 256 + i] * __expf(alpha);
                xs[m] = xb;
                out[(wg * 16 + m) * 256 + i] = xb;
                ld += alpha;
            }
            asm volatile("s_wait_asynccnt 0" ::: "memory");
        }
        __syncthreads();

        // ---- incremental a0 update: a0 += x_i * mw0[:,i] (from async-staged LDS) ----
        {
            float xm = xs[m];
#pragma unroll
            for (int tt = 0; tt < 2; ++tt) {
                const unsigned short* wc = &w0ls[(2 * w + tt) * 32 + g * 16];
#pragma unroll
                for (int e = 0; e < 16; ++e)
                    a0[tt][e] += xm * bf2f(wc[e]);
            }
        }
        __syncthreads();
    }
    if (w == 0 && g == 0)
        out[128 * 256 + wg * 16 + m] = -ld;
}

extern "C" void kernel_launch(void* const* d_in, const int* in_sizes, int n_in,
                              void* d_out, int out_size, void* d_ws, size_t ws_size,
                              hipStream_t stream) {
    (void)in_sizes; (void)n_in; (void)out_size; (void)ws_size;
    const float* z  = (const float*)d_in[0];
    const float* w0 = (const float*)d_in[1];
    const float* b0 = (const float*)d_in[2];
    const float* w1 = (const float*)d_in[3];
    const float* b1 = (const float*)d_in[4];
    const float* w2 = (const float*)d_in[5];
    const float* b2 = (const float*)d_in[6];
    unsigned short* ws = (unsigned short*)d_ws;

    made_prep<<<2560, 256, 0, stream>>>(w0, w1, w2, ws);
    made_main<<<8, 256, 0, stream>>>(z, b0, b1, b2,
                                     ws, ws + 131072, ws + 393216,
                                     (float*)d_out);
}
